// Multihead_Attention_origin_22780506538586
// MI455X (gfx1250) — matrix operand order repaired and checked by reading, not compiled, not run on hardware
//
#include <hip/hip_runtime.h>
#include <hip/hip_bf16.h>

// ---------------------------------------------------------------------------
// Multihead attention for MI455X (gfx1250), bf16 WMMA + f32 accumulation,
// with GLOBAL_LOAD_ASYNC_TO_LDS_B128 double-buffered tile streaming.
//   B=4, N=2048, EMB=768, HEADS=8, HEAD_DIM=96
// Pipeline:
//   1) qkv_proj_kernel : X(8192x768) @ {Wq,Wk,Wv} + bias -> Q/K/V bf16 [b][h][n][96]
//   2) attn_kernel     : flash-style streaming softmax(Q K^T) V  -> bf16 [b][n][768]
//   3) out_proj_kernel : attn @ Wo + bo -> f32 output
// ---------------------------------------------------------------------------

typedef __attribute__((ext_vector_type(16))) __bf16       v16bf;
typedef __attribute__((ext_vector_type(8)))  float        v8f;
typedef __attribute__((ext_vector_type(8)))  unsigned int v8u;
typedef int v4i __attribute__((vector_size(16)));  // matches builtin param type

#if defined(__AMDGCN__) && __has_builtin(__builtin_amdgcn_global_load_async_to_lds_b128) && \
    __has_builtin(__builtin_amdgcn_s_wait_asynccnt)
#define USE_ASYNC_LDS 1
#else
#define USE_ASYNC_LDS 0
#endif

// 16-byte global -> LDS copy. Async (ASYNCcnt-tracked, no VGPR round trip) on
// gfx1250; plain vector copy fallback elsewhere. Both addrs must be 16B aligned.
__device__ __forceinline__ void async_copy16(const void* g, void* lds) {
#if USE_ASYNC_LDS
  __builtin_amdgcn_global_load_async_to_lds_b128(
      (__attribute__((address_space(1))) v4i*)(uintptr_t)g,
      (__attribute__((address_space(3))) v4i*)(unsigned int)(uintptr_t)lds,
      /*offset=*/0, /*cpol=*/0);
#else
  *(uint4*)lds = *(const uint4*)g;
#endif
}
__device__ __forceinline__ void async_join() {
#if USE_ASYNC_LDS
  __builtin_amdgcn_s_wait_asynccnt(0);
#endif
}

__device__ __forceinline__ unsigned short f2bf(float f) {
  unsigned int u = __builtin_bit_cast(unsigned int, f);
  u = (u + 0x7FFFu + ((u >> 16) & 1u)) >> 16;  // RNE truncation to bf16
  return (unsigned short)u;
}

// A-matrix fragment (16x32 bf16, M x K): lane holds row (lane&15),
// kb = 8*(lane>=16); element e -> K = kb + (e<8 ? e : e+8).
__device__ __forceinline__ v16bf frag_a(const unsigned short* row, int kb) {
  v8u f;
#pragma unroll
  for (int j = 0; j < 8; ++j) {
    int k = kb + 2 * j + ((j >= 4) ? 8 : 0);
    f[j] = *(const unsigned int*)(row + k);
  }
  return __builtin_bit_cast(v16bf, f);
}

// B-matrix fragment (32x16 bf16, K x N): lane holds column N=(lane&15),
// kb = 8*(lane>=16); element e -> K = kb + (e<8 ? e : e+8) (row stride ld).
__device__ __forceinline__ v16bf frag_b(const unsigned short* p, int ld) {
  v8u f;
  const int kb = (threadIdx.x & 16) ? 8 : 0;
#pragma unroll
  for (int j = 0; j < 8; ++j) {
    int k = kb + 2 * j + ((j >= 4) ? 8 : 0);
    f[j] = (unsigned int)p[k * ld] | ((unsigned int)p[(k + 1) * ld] << 16);
  }
  return __builtin_bit_cast(v16bf, f);
}

__device__ __forceinline__ v8f wmma_bf16(v16bf a, v16bf b, v8f c) {
  return __builtin_amdgcn_wmma_f32_16x16x32_bf16(
      false, a, false, b, (short)0, c, false, false);
}

__device__ __forceinline__ float halfmax16(float v) {
#pragma unroll
  for (int m = 1; m < 16; m <<= 1) v = fmaxf(v, __shfl_xor(v, m, 32));
  return v;
}
__device__ __forceinline__ float halfsum16(float v) {
#pragma unroll
  for (int m = 1; m < 16; m <<= 1) v += __shfl_xor(v, m, 32);
  return v;
}

#define EMB 768
#define HEADS 8
#define HDIM 96
#define SEQ 2048
#define TOKS 8192  // B*N

// ---------------------------------------------------------------------------
// Kernel 1: fused QKV projection.  blockIdx.z selects Q/K/V.
// (f32 inputs need conversion on the way to LDS -> keep VGPR staging here.)
// ---------------------------------------------------------------------------
__global__ __launch_bounds__(256) void qkv_proj_kernel(
    const float* __restrict__ x,
    const float* __restrict__ Wq, const float* __restrict__ bq,
    const float* __restrict__ Wk, const float* __restrict__ bk,
    const float* __restrict__ Wv, const float* __restrict__ bv,
    unsigned short* __restrict__ Qo, unsigned short* __restrict__ Ko,
    unsigned short* __restrict__ Vo) {
  const float* W;
  const float* bias;
  unsigned short* out;
  if (blockIdx.z == 0) { W = Wq; bias = bq; out = Qo; }
  else if (blockIdx.z == 1) { W = Wk; bias = bk; out = Ko; }
  else { W = Wv; bias = bv; out = Vo; }

  __shared__ __align__(16) unsigned short At[128][36];
  __shared__ __align__(16) unsigned short Bt[32][132];

  const int tid = threadIdx.x;
  const int lane = tid & 31, wid = tid >> 5;
  const int wm = (wid & 3) * 32;
  const int wn = (wid >> 2) * 64;
  const int row0 = blockIdx.y * 128;
  const int col0 = blockIdx.x * 128;
  const int mrow = lane & 15;
  const int kb = (lane >= 16) ? 8 : 0;

  v8f acc[2][4] = {};

  for (int k0 = 0; k0 < EMB; k0 += 32) {
    __syncthreads();
#pragma unroll 4
    for (int i = tid; i < 128 * 32; i += 256) {
      int r = i >> 5, c = i & 31;
      At[r][c] = f2bf(x[(size_t)(row0 + r) * EMB + k0 + c]);
    }
#pragma unroll 4
    for (int i = tid; i < 32 * 128; i += 256) {
      int r = i >> 7, c = i & 127;
      Bt[r][c] = f2bf(W[(size_t)(k0 + r) * EMB + col0 + c]);
    }
    __syncthreads();

    v16bf a0 = frag_a(&At[wm + mrow][0], kb);
    v16bf a1 = frag_a(&At[wm + 16 + mrow][0], kb);
#pragma unroll
    for (int c = 0; c < 4; ++c) {
      v16bf b = frag_b(&Bt[0][wn + c * 16 + mrow], 132);
      acc[0][c] = wmma_bf16(a0, b, acc[0][c]);
      acc[1][c] = wmma_bf16(a1, b, acc[1][c]);
    }
  }

  const int hi8 = (lane >= 16) ? 8 : 0;
#pragma unroll
  for (int mi = 0; mi < 2; ++mi) {
#pragma unroll
    for (int c = 0; c < 4; ++c) {
      int coln = col0 + wn + c * 16 + mrow;
      float bia = bias[coln];
      int h = coln / HDIM, d = coln % HDIM;
#pragma unroll
      for (int r = 0; r < 8; ++r) {
        int m = r + hi8;
        int tok = row0 + wm + mi * 16 + m;  // 0..8191
        int bb = tok >> 11;
        int n = tok & 2047;
        float v = acc[mi][c][r] + bia;
        out[(((size_t)(bb * HEADS + h) * SEQ) + n) * HDIM + d] = f2bf(v);
      }
    }
  }
}

// ---------------------------------------------------------------------------
// Kernel 2: flash attention with async double-buffered K/V streaming.
// Workgroup = 128 query rows of one (b,h); 8 waves x 16 rows; key blocks of 32.
// Reference quirk: softmax(unscaled logits)/sqrt(96) folded into the epilogue.
// ---------------------------------------------------------------------------
__global__ __launch_bounds__(256) void attn_kernel(
    const unsigned short* __restrict__ Q,
    const unsigned short* __restrict__ K,
    const unsigned short* __restrict__ V,
    unsigned short* __restrict__ O) {  // bf16 [b][n][768] heads interleaved
  // rows padded to multiples of 16B for async b128 writes; strides chosen
  // (52 / 18 words) to keep fragment reads bank-conflict free.
  __shared__ __align__(16) unsigned short Qs[128][104];
  __shared__ __align__(16) unsigned short Kt[2][96][36];   // K^T: [d][key]
  __shared__ __align__(16) unsigned short Vs[2][32][104];
  __shared__ __align__(16) unsigned short Ps[8][16][36];   // per-wave P scratch

  const int tid = threadIdx.x, lane = tid & 31, wid = tid >> 5;
  const int bh = blockIdx.y;  // 0..31
  const int b = bh >> 3, h = bh & 7;
  const int q0 = blockIdx.x * 128;
  const size_t base = (size_t)bh * SEQ * HDIM;

  // stage V(buf)/K^T(buf) for key block kblk: V via async b128,
  // K via VGPR transpose (issued in the same slot so it overlaps compute too)
  auto stage_kv = [&](int buf, int kblk) {
    for (int i = tid; i < 32 * 12; i += 256) {        // 384 x 16B
      int kr = i / 12, ch = i % 12;
      async_copy16(&V[base + (size_t)(kblk + kr) * HDIM + ch * 8],
                   &Vs[buf][kr][ch * 8]);
    }
#pragma unroll 4
    for (int i = tid; i < 32 * HDIM; i += 256) {      // transpose K -> Kt
      int kr = i / HDIM, d = i % HDIM;
      Kt[buf][d][kr] = K[base + (size_t)(kblk + kr) * HDIM + d];
    }
  };

  // stage Q tile via async b128 (1536 x 16B)
  for (int i = tid; i < 128 * 12; i += 256) {
    int r = i / 12, ch = i % 12;
    async_copy16(&Q[base + (size_t)(q0 + r) * HDIM + ch * 8], &Qs[r][ch * 8]);
  }
  stage_kv(0, 0);
  async_join();
  __syncthreads();

  const int mrow = lane & 15;
  const int kb = (lane >= 16) ? 8 : 0;
  const int hi8 = (lane >= 16) ? 8 : 0;

  v16bf qa[3];
#pragma unroll
  for (int dc = 0; dc < 3; ++dc)
    qa[dc] = frag_a(&Qs[wid * 16 + mrow][dc * 32], kb);

  v8f acc[6] = {};
  float mmax[8], lsum[8];
#pragma unroll
  for (int r = 0; r < 8; ++r) { mmax[r] = -1e30f; lsum[r] = 0.f; }

  for (int kblk = 0; kblk < SEQ; kblk += 32) {
    const int cur = (kblk >> 5) & 1;
    const bool more = (kblk + 32) < SEQ;
    if (more) stage_kv(cur ^ 1, kblk + 32);  // prefetch next block (async)

    // S = Q K^T for this 16q x 32key block (two 16x16 halves)
    v8f s0 = {}, s1 = {};
#pragma unroll
    for (int dc = 0; dc < 3; ++dc) {
      v16bf bk0 = frag_b(&Kt[cur][dc * 32][mrow], 36);
      v16bf bk1 = frag_b(&Kt[cur][dc * 32][16 + mrow], 36);
      s0 = wmma_bf16(qa[dc], bk0, s0);
      s1 = wmma_bf16(qa[dc], bk1, s1);
    }

    // online softmax; element r covers rows r (lanes 0-15) / r+8 (lanes 16-31)
#pragma unroll
    for (int r = 0; r < 8; ++r) {
      float blkmax = halfmax16(fmaxf(s0[r], s1[r]));
      float mnew = fmaxf(mmax[r], blkmax);
      float p0 = __expf(s0[r] - mnew);
      float p1 = __expf(s1[r] - mnew);
      float scale = __expf(mmax[r] - mnew);
      lsum[r] = lsum[r] * scale + halfsum16(p0 + p1);
      mmax[r] = mnew;
#pragma unroll
      for (int c = 0; c < 6; ++c) acc[c][r] *= scale;
      int m = r + hi8;
      Ps[wid][m][mrow] = f2bf(p0);       // keys 0..15
      Ps[wid][m][16 + mrow] = f2bf(p1);  // keys 16..31
    }

    // O += P V  (per-wave LDS round trip re-lays P into A-fragment order;
    // DS ops are in-order within a wave)
    v16bf pa = frag_a(&Ps[wid][mrow][0], kb);
#pragma unroll
    for (int c = 0; c < 6; ++c) {
      v16bf bv = frag_b(&Vs[cur][0][c * 16 + mrow], 104);
      acc[c] = wmma_bf16(pa, bv, acc[c]);
    }

    if (more) {
      async_join();     // own async transfers for next block landed
      __syncthreads();  // everyone's staging of next block complete
    }
  }

  // epilogue: O / (l * sqrt(96)), write bf16 [b][n][h*96 + d]
  const float inv_sqrt_d = 0.10206207261596575f;  // 1/sqrt(96)
#pragma unroll
  for (int r = 0; r < 8; ++r) {
    float s = inv_sqrt_d / lsum[r];
    int m = r + hi8;
    int tok = q0 + wid * 16 + m;
    size_t orow = ((size_t)b * SEQ + tok) * EMB + h * HDIM;
#pragma unroll
    for (int c = 0; c < 6; ++c)
      O[orow + c * 16 + mrow] = f2bf(acc[c][r] * s);
  }
}

// ---------------------------------------------------------------------------
// Kernel 3: output projection (bf16 A async-staged, f32 W/bias -> f32 out)
// ---------------------------------------------------------------------------
__global__ __launch_bounds__(256) void out_proj_kernel(
    const unsigned short* __restrict__ A,  // bf16 [8192][768]
    const float* __restrict__ W, const float* __restrict__ bias,
    float* __restrict__ out) {
  __shared__ __align__(16) unsigned short At[128][40];   // 80B rows (16B mult)
  __shared__ __align__(16) unsigned short Bt[32][132];

  const int tid = threadIdx.x;
  const int lane = tid & 31, wid = tid >> 5;
  const int wm = (wid & 3) * 32;
  const int wn = (wid >> 2) * 64;
  const int row0 = blockIdx.y * 128;
  const int col0 = blockIdx.x * 128;
  const int mrow = lane & 15;
  const int kb = (lane >= 16) ? 8 : 0;

  v8f acc[2][4] = {};

  for (int k0 = 0; k0 < EMB; k0 += 32) {
    __syncthreads();
    // A tile: 128x32 bf16 = 512 x 16B chunks, async to LDS
    for (int i = tid; i < 128 * 4; i += 256) {
      int r = i >> 2, ch = i & 3;
      async_copy16(&A[(size_t)(row0 + r) * EMB + k0 + ch * 8], &At[r][ch * 8]);
    }
#pragma unroll 4
    for (int i = tid; i < 32 * 128; i += 256) {
      int r = i >> 7, c = i & 127;
      Bt[r][c] = f2bf(W[(size_t)(k0 + r) * EMB + col0 + c]);
    }
    async_join();
    __syncthreads();

    v16bf a0 = frag_a(&At[wm + mrow][0], kb);
    v16bf a1 = frag_a(&At[wm + 16 + mrow][0], kb);
#pragma unroll
    for (int c = 0; c < 4; ++c) {
      v16bf b = frag_b(&Bt[0][wn + c * 16 + mrow], 132);
      acc[0][c] = wmma_bf16(a0, b, acc[0][c]);
      acc[1][c] = wmma_bf16(a1, b, acc[1][c]);
    }
  }

  const int hi8 = (lane >= 16) ? 8 : 0;
#pragma unroll
  for (int mi = 0; mi < 2; ++mi) {
#pragma unroll
    for (int c = 0; c < 4; ++c) {
      int coln = col0 + wn + c * 16 + mrow;
      float bia = bias[coln];
#pragma unroll
      for (int r = 0; r < 8; ++r) {
        int m = r + hi8;
        int tok = row0 + wm + mi * 16 + m;
        out[(size_t)tok * EMB + coln] = acc[mi][c][r] + bia;
      }
    }
  }
}

// ---------------------------------------------------------------------------
extern "C" void kernel_launch(void* const* d_in, const int* in_sizes, int n_in,
                              void* d_out, int out_size, void* d_ws,
                              size_t ws_size, hipStream_t stream) {
  const float* x  = (const float*)d_in[0];
  const float* Wq = (const float*)d_in[1];
  const float* bq = (const float*)d_in[2];
  const float* Wk = (const float*)d_in[3];
  const float* bk = (const float*)d_in[4];
  const float* Wv = (const float*)d_in[5];
  const float* bv = (const float*)d_in[6];
  const float* Wo = (const float*)d_in[7];
  const float* bo = (const float*)d_in[8];
  float* out = (float*)d_out;

  const size_t qkv_bytes =
      (size_t)4 * HEADS * SEQ * HDIM * sizeof(unsigned short);  // 12,582,912
  char* ws = (char*)d_ws;
  unsigned short* Qb = (unsigned short*)(ws + 0 * qkv_bytes);
  unsigned short* Kb = (unsigned short*)(ws + 1 * qkv_bytes);
  unsigned short* Vb = (unsigned short*)(ws + 2 * qkv_bytes);
  unsigned short* Ab = (unsigned short*)(ws + 3 * qkv_bytes);  // bf16 [8192][768]

  qkv_proj_kernel<<<dim3(EMB / 128, TOKS / 128, 3), 256, 0, stream>>>(
      x, Wq, bq, Wk, bk, Wv, bv, Qb, Kb, Vb);
  attn_kernel<<<dim3(SEQ / 128, 4 * HEADS), 256, 0, stream>>>(Qb, Kb, Vb, Ab);
  out_proj_kernel<<<dim3(EMB / 128, TOKS / 128), 256, 0, stream>>>(Ab, Wo, bo, out);
}
